// MoETransformersBlock_14276471291958
// MI455X (gfx1250) — matrix operand order repaired and checked by reading, not compiled, not run on hardware
//
#include <hip/hip_runtime.h>
#include <hip/hip_bf16.h>

// ---------------------------------------------------------------------------
// MoE transformer block for MI455X (gfx1250), bf16 WMMA everywhere.
//   B=2 S=2048 EMB=1024 H=16 KV=4 D=128 E=8 TOPK=2 MOE_HID=1024, T=4096 tokens
// ---------------------------------------------------------------------------

#define EMB 1024
#define NH 16
#define NKV 4
#define HD 128
#define NEXP 8
#define MOEH 1024
#define BB 2
#define SS 2048
#define TT 4096            // B*S
#define QKV_N 3072         // 2048 Q + 512 K + 512 V

typedef __attribute__((ext_vector_type(16))) __bf16 v16bf;
typedef __attribute__((ext_vector_type(8)))  float  v8f;
typedef __attribute__((ext_vector_type(4)))  unsigned int u32x4;
typedef __attribute__((ext_vector_type(8)))  int  i32x8;
typedef __attribute__((ext_vector_type(4)))  int  i32x4;

#ifndef __has_builtin
#define __has_builtin(x) 0
#endif
#if __has_builtin(__builtin_amdgcn_tensor_load_to_lds) && \
    __has_builtin(__builtin_amdgcn_s_wait_tensorcnt)
#define HAVE_TDM 1
#else
#define HAVE_TDM 0
#endif

union Frag { struct { u32x4 lo, hi; } s; v16bf v; };

__device__ __forceinline__ unsigned short f2bf(float f) {
  unsigned int u = __float_as_uint(f);
  u += 0x7fffu + ((u >> 16) & 1u);      // round-to-nearest-even
  return (unsigned short)(u >> 16);
}

__device__ __forceinline__ v8f wmma_bf16(v16bf a, v16bf b, v8f c) {
  return __builtin_amdgcn_wmma_f32_16x16x32_bf16(
      false, a, false, b, (short)0, c, false, false);
}

// A-fragment (16x32 bf16, row-major source, one row per lane).
__device__ __forceinline__ v16bf load_a_frag(const unsigned short* A, int lda, int lane) {
  int half = lane >> 4, r = lane & 15;
  const unsigned short* p = A + (size_t)r * lda + half * 8;
  Frag f;
  f.s.lo = *(const u32x4*)(p);
  f.s.hi = *(const u32x4*)(p + 16);
  return f.v;
}

// Same but per-lane row pointer already resolved (MoE token gather).
__device__ __forceinline__ v16bf load_a_frag_ptr(const unsigned short* rowp, int half) {
  const unsigned short* p = rowp + half * 8;
  Frag f;
  f.s.lo = *(const u32x4*)(p);
  f.s.hi = *(const u32x4*)(p + 16);
  return f.v;
}

// B-fragment (32x16 bf16) from row-major W[N,K] (computes h @ W^T).
__device__ __forceinline__ v16bf load_b_frag(const unsigned short* W, int ldb, int lane) {
  int half = lane >> 4, c = lane & 15;
  const unsigned short* p = W + (size_t)c * ldb + half * 8;
  Frag f;
  f.s.lo = *(const u32x4*)(p);
  f.s.hi = *(const u32x4*)(p + 16);
  return f.v;
}

#if HAVE_TDM
// Issue a TDM 2D tile load: tile_rows x tile_cols bf16 elements, row stride
// row_stride_elems, packed contiguously into LDS at lds_byte_addr.
// D# packing per CDNA5 ISA.
// amdgpu-toolchain (clang-23) 6-arg builtin form.
__device__ __forceinline__ void tdm_load_tile_2d(unsigned lds_byte_addr,
                                                 const void* gptr,
                                                 unsigned tile_cols, unsigned tile_rows,
                                                 unsigned long long row_stride_elems) {
  unsigned long long ga = (unsigned long long)(uintptr_t)gptr;
  u32x4 g0;
  g0[0] = 1u;                                           // count=1, user D#
  g0[1] = lds_byte_addr;                                // lds_addr [63:32]
  g0[2] = (unsigned)(ga & 0xffffffffu);                 // global_addr lo
  g0[3] = (unsigned)((ga >> 32) & 0x01ffffffu) | 0x80000000u; // addr hi | type=2
  i32x8 g1;
  g1[0] = 0x00010000;                                   // data_size=1 (2B/elem)
  g1[1] = (int)((tile_cols & 0xffffu) << 16);           // tensor_dim0[15:0]
  g1[2] = (int)(((tile_cols >> 16) & 0xffffu) | ((tile_rows & 0xffffu) << 16));
  g1[3] = (int)(((tile_rows >> 16) & 0xffffu) | ((tile_cols & 0xffffu) << 16)); // tile_dim0
  g1[4] = (int)(tile_rows & 0xffffu);                   // tile_dim1 (tile_dim2=0)
  g1[5] = (int)(row_stride_elems & 0xffffffffu);        // tensor_dim0_stride lo
  g1[6] = (int)((row_stride_elems >> 32) & 0xffffu);    // stride hi
  g1[7] = 0;
  i32x4 z4 = {0, 0, 0, 0};
  i32x8 z8 = {0, 0, 0, 0, 0, 0, 0, 0};
  __builtin_amdgcn_tensor_load_to_lds(g0, g1, z4, z4, z8, 0);
}
#define WAIT_TENSOR(n) __builtin_amdgcn_s_wait_tensorcnt(n)
#endif

// ---------------------------------------------------------------------------
// fp32 -> bf16 bulk convert
// ---------------------------------------------------------------------------
__global__ void __launch_bounds__(256)
cvt_f32_bf16(const float* __restrict__ src, unsigned short* __restrict__ dst, int n) {
  int i = blockIdx.x * 256 + threadIdx.x;
  if (i < n) dst[i] = f2bf(src[i]);
}

// ---------------------------------------------------------------------------
// rmsnorm over EMB, fp32 in -> bf16 out, one block per token
// ---------------------------------------------------------------------------
__global__ void __launch_bounds__(256)
rmsnorm_kernel(const float* __restrict__ x, const float* __restrict__ w,
               unsigned short* __restrict__ out) {
  int t = blockIdx.x;
  const float* xr = x + (size_t)t * EMB;
  float v[4]; float ss = 0.f;
#pragma unroll
  for (int i = 0; i < 4; ++i) { v[i] = xr[threadIdx.x + 256 * i]; ss += v[i] * v[i]; }
  for (int m = 16; m >= 1; m >>= 1) ss += __shfl_xor(ss, m, 32);
  __shared__ float sred[8];
  if ((threadIdx.x & 31) == 0) sred[threadIdx.x >> 5] = ss;
  __syncthreads();
  float tot = 0.f;
#pragma unroll
  for (int i = 0; i < 8; ++i) tot += sred[i];
  float rstd = rsqrtf(tot * (1.0f / EMB) + 1e-6f);
#pragma unroll
  for (int i = 0; i < 4; ++i) {
    int c = threadIdx.x + 256 * i;
    out[(size_t)t * EMB + c] = f2bf(v[i] * rstd * w[c]);
  }
}

// ---------------------------------------------------------------------------
// Register-blocked bf16 GEMM: C[M,N] = A[M,K] @ W[N,K]^T (+resid).
// Each wave computes a (16*TM) x (16*TN) tile: TM*TN f32 accumulators,
// TM+TN fragment loads per 32-K step -> ~21 FLOP/byte at 4x2.
// ---------------------------------------------------------------------------
template <int TM, int TN>
__global__ void __launch_bounds__(256)
gemm_rb_kernel(const unsigned short* __restrict__ A, int lda,
               const unsigned short* __restrict__ W, int ldb,
               float* __restrict__ C, int ldc,
               int M, int N, int K, const float* __restrict__ resid) {
  int lane = threadIdx.x & 31;
  int wid = blockIdx.x * 8 + (threadIdx.x >> 5);
  int tn = N / (16 * TN);
  int tiles = (M / (16 * TM)) * tn;
  if (wid >= tiles) return;
  int mt = wid / tn, nt = wid - mt * tn;
  const unsigned short* Ab = A + (size_t)(mt * 16 * TM) * lda;
  const unsigned short* Wb = W + (size_t)(nt * 16 * TN) * ldb;
  v8f acc[TM][TN] = {};
  for (int k = 0; k < K; k += 32) {
    v16bf a[TM], b[TN];
#pragma unroll
    for (int i = 0; i < TM; ++i)
      a[i] = load_a_frag(Ab + (size_t)i * 16 * lda + k, lda, lane);
#pragma unroll
    for (int j = 0; j < TN; ++j)
      b[j] = load_b_frag(Wb + (size_t)j * 16 * ldb + k, ldb, lane);
#pragma unroll
    for (int i = 0; i < TM; ++i)
#pragma unroll
      for (int j = 0; j < TN; ++j)
        acc[i][j] = wmma_bf16(a[i], b[j], acc[i][j]);
  }
  int r = lane & 15, halfo = (lane >> 4) << 3;
#pragma unroll
  for (int i = 0; i < TM; ++i) {
#pragma unroll
    for (int j = 0; j < TN; ++j) {
#pragma unroll
      for (int jj = 0; jj < 8; ++jj) {
        size_t idx = (size_t)(mt * 16 * TM + i * 16 + jj + halfo) * ldc
                   + (nt * 16 * TN + j * 16 + r);
        float v = acc[i][j][jj];
        if (resid) v += resid[idx];
        C[idx] = v;
      }
    }
  }
}

// ---------------------------------------------------------------------------
// Q/K head rmsnorm + RoPE; fp32 QKV gemm output -> bf16 [b,h,s,d] rows.
// ---------------------------------------------------------------------------
__global__ void __launch_bounds__(256)
qknorm_rope_kernel(const float* __restrict__ qkv, int colofs,
                   const float* __restrict__ nw, const int* __restrict__ pos_ids,
                   unsigned short* __restrict__ out, int nheads) {
  int lane = threadIdx.x & 31;
  int wid = blockIdx.x * (blockDim.x >> 5) + (threadIdx.x >> 5);
  if (wid >= TT * nheads) return;
  int t = wid / nheads, h = wid - t * nheads;
  int b = t / SS, s = t - b * SS;
  const float* src = qkv + (size_t)t * QKV_N + colofs + h * HD;
  float v0 = src[lane], v1 = src[lane + 32], v2 = src[lane + 64], v3 = src[lane + 96];
  float ssq = v0 * v0 + v1 * v1 + v2 * v2 + v3 * v3;
  for (int m = 16; m >= 1; m >>= 1) ssq += __shfl_xor(ssq, m, 32);
  float rstd = rsqrtf(ssq * (1.0f / HD) + 1e-6f);
  v0 *= rstd * nw[lane];      v1 *= rstd * nw[lane + 32];
  v2 *= rstd * nw[lane + 64]; v3 *= rstd * nw[lane + 96];
  float pos = (float)pos_ids[s];
  float f0 = pos * __expf(-9.210340371976184f * (float)lane / 64.f);
  float f1 = pos * __expf(-9.210340371976184f * (float)(lane + 32) / 64.f);
  float c0 = __cosf(f0), s0 = __sinf(f0), c1 = __cosf(f1), s1 = __sinf(f1);
  unsigned short* dst = out + (((size_t)(b * nheads + h) * SS + s) << 7);
  dst[lane]      = f2bf(v0 * c0 - v2 * s0);
  dst[lane + 32] = f2bf(v1 * c1 - v3 * s1);
  dst[lane + 64] = f2bf(v2 * c0 + v0 * s0);
  dst[lane + 96] = f2bf(v3 * c1 + v1 * s1);
}

// V: fp32 -> bf16, transposed to [b, kv, d, s]
__global__ void __launch_bounds__(256)
v_transpose_kernel(const float* __restrict__ qkv, unsigned short* __restrict__ Vt) {
  int i = blockIdx.x * 256 + threadIdx.x;
  if (i >= TT * NKV * HD) return;
  int t = i >> 9, c = i & 511;
  int b = t / SS, s = t - b * SS;
  int kh = c >> 7, d = c & 127;
  Vt[(((size_t)(b * NKV + kh) * HD + d) * SS + s)] =
      f2bf(qkv[(size_t)t * QKV_N + 2560 + c]);
}

// ---------------------------------------------------------------------------
// Flash attention: one wave per (b, h, 16-row q-tile); online softmax;
// 32-key steps; P staged through LDS to convert C-layout -> A-layout.
// ---------------------------------------------------------------------------
__global__ void __launch_bounds__(32)
attn_kernel(const unsigned short* __restrict__ Qr, const unsigned short* __restrict__ Kr,
            const unsigned short* __restrict__ Vt, unsigned short* __restrict__ ctx) {
  __shared__ __align__(16) unsigned short P[16 * 32];
  int lane = threadIdx.x;
  int half = lane >> 4, lr = lane & 15;
  int qtiles = SS >> 4;
  int qt = blockIdx.x & (qtiles - 1);
  int bh = blockIdx.x / qtiles;
  int b = bh / NH, h = bh - b * NH;
  int kvh = h / (NH / NKV);
  const unsigned short* Qb = Qr + (((size_t)(b * NH + h) * SS) + qt * 16) * HD;
  const unsigned short* Kb = Kr + ((size_t)(b * NKV + kvh) * SS) * HD;
  const unsigned short* Vb = Vt + ((size_t)(b * NKV + kvh) * HD) * SS;

  v16bf qa[4];
#pragma unroll
  for (int dt = 0; dt < 4; ++dt) qa[dt] = load_a_frag(Qb + dt * 32, HD, lane);

  float rmax[8], rsum[8];
  v8f oacc[8] = {};
#pragma unroll
  for (int j = 0; j < 8; ++j) { rmax[j] = -1e30f; rsum[j] = 0.f; }
  const float scale = 0.08838834764831845f;   // 1/sqrt(128)

  int ktend = (qt * 16 + 15) >> 5;
  for (int kt = 0; kt <= ktend; ++kt) {
    int kbase = kt * 32;
    v8f s0 = {}, s1 = {};
#pragma unroll
    for (int dt = 0; dt < 4; ++dt) {
      v16bf b0 = load_b_frag(Kb + (size_t)kbase * HD + dt * 32, HD, lane);
      v16bf b1 = load_b_frag(Kb + (size_t)(kbase + 16) * HD + dt * 32, HD, lane);
      s0 = wmma_bf16(qa[dt], b0, s0);
      s1 = wmma_bf16(qa[dt], b1, s1);
    }
    int key0 = kbase + lr, key1 = kbase + 16 + lr;
#pragma unroll
    for (int j = 0; j < 8; ++j) {
      int m = qt * 16 + j + 8 * half;
      float x0 = (key0 <= m) ? s0[j] * scale : -1e30f;
      float x1 = (key1 <= m) ? s1[j] * scale : -1e30f;
      float tmax = fmaxf(x0, x1);
      for (int d = 8; d >= 1; d >>= 1) tmax = fmaxf(tmax, __shfl_xor(tmax, d, 32));
      float nmax = fmaxf(rmax[j], tmax);
      float sf = __expf(rmax[j] - nmax);
      float p0 = __expf(x0 - nmax), p1 = __expf(x1 - nmax);
      float ps = p0 + p1;
      for (int d = 8; d >= 1; d >>= 1) ps += __shfl_xor(ps, d, 32);
      rsum[j] = rsum[j] * sf + ps;
      rmax[j] = nmax;
#pragma unroll
      for (int dt8 = 0; dt8 < 8; ++dt8) oacc[dt8][j] *= sf;
      P[(j + 8 * half) * 32 + lr]      = f2bf(p0);
      P[(j + 8 * half) * 32 + lr + 16] = f2bf(p1);
    }
    __syncthreads();
    v16bf pa = load_a_frag(P, 32, lane);
    __syncthreads();
#pragma unroll
    for (int dt8 = 0; dt8 < 8; ++dt8) {
      v16bf bv = load_b_frag(Vb + (size_t)(dt8 * 16) * SS + kbase, SS, lane);
      oacc[dt8] = wmma_bf16(pa, bv, oacc[dt8]);
    }
  }
#pragma unroll
  for (int j = 0; j < 8; ++j) {
    int m = qt * 16 + j + 8 * half;
    float inv = 1.f / rsum[j];
    unsigned short* dst = ctx + ((size_t)(b * SS + m)) * (NH * HD) + h * HD;
#pragma unroll
    for (int dt8 = 0; dt8 < 8; ++dt8)
      dst[dt8 * 16 + lr] = f2bf(oacc[dt8][j] * inv);
  }
}

// ---------------------------------------------------------------------------
// rmsnorm2 + router (softmax top-2, renormalized) + per-expert token lists
// ---------------------------------------------------------------------------
__global__ void zero_counts(int* cnt) { if (threadIdx.x < NEXP) cnt[threadIdx.x] = 0; }

__global__ void __launch_bounds__(256)
rms2_router_kernel(const float* __restrict__ x, const float* __restrict__ w,
                   const float* __restrict__ rw, unsigned short* __restrict__ h2,
                   int* __restrict__ cnt, int* __restrict__ tok, float* __restrict__ wtl) {
  int t = blockIdx.x;
  const float* xr = x + (size_t)t * EMB;
  float v[4]; float ssq = 0.f;
#pragma unroll
  for (int i = 0; i < 4; ++i) { v[i] = xr[threadIdx.x + 256 * i]; ssq += v[i] * v[i]; }
  for (int m = 16; m >= 1; m >>= 1) ssq += __shfl_xor(ssq, m, 32);
  __shared__ float sred[8];
  __shared__ float slog[NEXP];
  if (threadIdx.x < NEXP) slog[threadIdx.x] = 0.f;
  if ((threadIdx.x & 31) == 0) sred[threadIdx.x >> 5] = ssq;
  __syncthreads();
  float tot = 0.f;
#pragma unroll
  for (int i = 0; i < 8; ++i) tot += sred[i];
  float rstd = rsqrtf(tot * (1.0f / EMB) + 1e-6f);
  float p[NEXP];
#pragma unroll
  for (int e = 0; e < NEXP; ++e) p[e] = 0.f;
#pragma unroll
  for (int i = 0; i < 4; ++i) {
    int c = threadIdx.x + 256 * i;
    float hv = v[i] * rstd * w[c];
    h2[(size_t)t * EMB + c] = f2bf(hv);
#pragma unroll
    for (int e = 0; e < NEXP; ++e) p[e] += hv * rw[e * EMB + c];
  }
#pragma unroll
  for (int e = 0; e < NEXP; ++e) {
    for (int m = 16; m >= 1; m >>= 1) p[e] += __shfl_xor(p[e], m, 32);
    if ((threadIdx.x & 31) == 0) atomicAdd(&slog[e], p[e]);
  }
  __syncthreads();
  if (threadIdx.x == 0) {
    float mx = -1e30f;
    for (int e = 0; e < NEXP; ++e) mx = fmaxf(mx, slog[e]);
    float ex[NEXP];
    for (int e = 0; e < NEXP; ++e) ex[e] = __expf(slog[e] - mx);
    int i0 = 0;
    for (int e = 1; e < NEXP; ++e) if (ex[e] > ex[i0]) i0 = e;
    int i1 = (i0 == 0) ? 1 : 0;
    for (int e = 0; e < NEXP; ++e) { if (e == i0) continue; if (ex[e] > ex[i1]) i1 = e; }
    float w0 = ex[i0] / (ex[i0] + ex[i1]);
    float w1 = 1.f - w0;
    int s0 = atomicAdd(&cnt[i0], 1); tok[i0 * TT + s0] = t; wtl[i0 * TT + s0] = w0;
    int s1 = atomicAdd(&cnt[i1], 1); tok[i1 * TT + s1] = t; wtl[i1 * TT + s1] = w1;
  }
}

// ---------------------------------------------------------------------------
// MoE gate_up GEMM with token gather + SiLU*up epilogue -> act (bf16)
// wave: (e, 32-slot tile mt, 16-col tile nt) -> gate 32x16 and up 32x16;
// A fragments reused across gate and up WMMAs.
// ---------------------------------------------------------------------------
__global__ void __launch_bounds__(256)
moe_gateup_kernel(const unsigned short* __restrict__ h2,
                  const unsigned short* __restrict__ wgu,
                  const int* __restrict__ cnt, const int* __restrict__ tok,
                  unsigned short* __restrict__ act) {
  int lane = threadIdx.x & 31;
  int half = lane >> 4, r = lane & 15;
  int wid = blockIdx.x * 8 + (threadIdx.x >> 5);
  int nt = wid & 63;                 // 64 col tiles of 16
  int mt = (wid >> 6) & 127;         // 128 row tiles of 32
  int e  = wid >> 13;
  if (e >= NEXP) return;
  int c = cnt[e];
  if (mt * 32 >= c) return;
  int sr0 = mt * 32 + r, sr1 = sr0 + 16;
  int t0 = tok[e * TT + (sr0 < c ? sr0 : c - 1)];
  int t1 = tok[e * TT + (sr1 < c ? sr1 : c - 1)];
  const unsigned short* a0row = h2 + (size_t)t0 * EMB;
  const unsigned short* a1row = h2 + (size_t)t1 * EMB;
  const unsigned short* wg = wgu + (size_t)e * (2 * MOEH) * EMB + (size_t)(nt * 16) * EMB;
  const unsigned short* wu = wg + (size_t)MOEH * EMB;
  v8f g0 = {}, g1 = {}, u0 = {}, u1 = {};
  for (int k = 0; k < EMB; k += 32) {
    v16bf a0 = load_a_frag_ptr(a0row + k, half);
    v16bf a1 = load_a_frag_ptr(a1row + k, half);
    v16bf bg = load_b_frag(wg + k, EMB, lane);
    v16bf bu = load_b_frag(wu + k, EMB, lane);
    g0 = wmma_bf16(a0, bg, g0);
    g1 = wmma_bf16(a1, bg, g1);
    u0 = wmma_bf16(a0, bu, u0);
    u1 = wmma_bf16(a1, bu, u1);
  }
  unsigned short* dst = act + ((size_t)e * TT + mt * 32) * MOEH + nt * 16;
#pragma unroll
  for (int jj = 0; jj < 8; ++jj) {
    int m = jj + 8 * half;
    float ga = g0[jj], ua = u0[jj];
    dst[(size_t)m * MOEH + r] = f2bf(ga / (1.f + __expf(-ga)) * ua);
    ga = g1[jj]; ua = u1[jj];
    dst[(size_t)(m + 16) * MOEH + r] = f2bf(ga / (1.f + __expf(-ga)) * ua);
  }
}

// ---------------------------------------------------------------------------
// MoE down GEMM (32x32 tile/wave) + weighted atomic scatter onto d_out.
// B tiles (32 rows x 32 k) are DMA'd into LDS by the Tensor Data Mover,
// double-buffered, pipelined with s_wait_tensorcnt.
// ---------------------------------------------------------------------------
__global__ void __launch_bounds__(256)
moe_down_kernel(const unsigned short* __restrict__ act,
                const unsigned short* __restrict__ wdn,
                const int* __restrict__ cnt, const int* __restrict__ tok,
                const float* __restrict__ wtl, float* __restrict__ out) {
#if HAVE_TDM
  __shared__ __align__(16) unsigned short btile[8][2][32 * 32];   // 32 KiB
#endif
  int lane = threadIdx.x & 31;
  int half = lane >> 4, r = lane & 15;
  int wv = __builtin_amdgcn_readfirstlane((int)(threadIdx.x >> 5));
  int wid = blockIdx.x * 8 + wv;
  int nt = wid & 31;                 // 32 col tiles of 32
  int mt = (wid >> 5) & 127;         // 128 slot tiles of 32
  int e  = wid >> 12;
  if (e >= NEXP) return;
  int c = cnt[e];
  if (mt * 32 >= c) return;
  const unsigned short* Ab = act + ((size_t)e * TT + mt * 32) * MOEH;
  const unsigned short* Wb = wdn + (size_t)e * EMB * MOEH + (size_t)(nt * 32) * MOEH;
  v8f acc[2][2] = {};
#if HAVE_TDM
  unsigned lds0 = (unsigned)(uintptr_t)&btile[wv][0][0];
  unsigned lds1 = (unsigned)(uintptr_t)&btile[wv][1][0];
  tdm_load_tile_2d(lds0, Wb, 32, 32, (unsigned long long)MOEH);
  int cur = 0;
  for (int k = 0; k < MOEH; k += 32) {
    if (k + 32 < MOEH) {
      tdm_load_tile_2d(cur ? lds0 : lds1, Wb + k + 32, 32, 32,
                       (unsigned long long)MOEH);
      WAIT_TENSOR(1);                // oldest (current buffer) done
    } else {
      WAIT_TENSOR(0);
    }
    const unsigned short* bt = &btile[wv][cur][0];
    v16bf b0 = load_b_frag(bt, 32, lane);
    v16bf b1 = load_b_frag(bt + 16 * 32, 32, lane);
    v16bf a0 = load_a_frag(Ab + k, MOEH, lane);
    v16bf a1 = load_a_frag(Ab + (size_t)16 * MOEH + k, MOEH, lane);
    acc[0][0] = wmma_bf16(a0, b0, acc[0][0]);
    acc[0][1] = wmma_bf16(a0, b1, acc[0][1]);
    acc[1][0] = wmma_bf16(a1, b0, acc[1][0]);
    acc[1][1] = wmma_bf16(a1, b1, acc[1][1]);
    cur ^= 1;
  }
#else
  for (int k = 0; k < MOEH; k += 32) {
    v16bf b0 = load_b_frag(Wb + k, MOEH, lane);
    v16bf b1 = load_b_frag(Wb + (size_t)16 * MOEH + k, MOEH, lane);
    v16bf a0 = load_a_frag(Ab + k, MOEH, lane);
    v16bf a1 = load_a_frag(Ab + (size_t)16 * MOEH + k, MOEH, lane);
    acc[0][0] = wmma_bf16(a0, b0, acc[0][0]);
    acc[0][1] = wmma_bf16(a0, b1, acc[0][1]);
    acc[1][0] = wmma_bf16(a1, b0, acc[1][0]);
    acc[1][1] = wmma_bf16(a1, b1, acc[1][1]);
  }
#endif
#pragma unroll
  for (int i = 0; i < 2; ++i) {
#pragma unroll
    for (int jj = 0; jj < 8; ++jj) {
      int s = mt * 32 + i * 16 + jj + 8 * half;
      if (s < c) {
        int t = tok[e * TT + s];
        float wgt = wtl[e * TT + s];
#pragma unroll
        for (int j = 0; j < 2; ++j)
          atomicAdd(&out[(size_t)t * EMB + nt * 32 + j * 16 + r],
                    wgt * acc[i][j][jj]);
      }
    }
  }
}

// ---------------------------------------------------------------------------
// Host side
// ---------------------------------------------------------------------------
extern "C" void kernel_launch(void* const* d_in, const int* in_sizes, int n_in,
                              void* d_out, int out_size, void* d_ws, size_t ws_size,
                              hipStream_t stream) {
  const float* x         = (const float*)d_in[0];
  const int*   pos_ids   = (const int*)d_in[1];
  // d_in[2] = attn_mask (causal; implemented analytically)
  const float* norm1_w   = (const float*)d_in[3];
  const float* norm2_w   = (const float*)d_in[4];
  const float* qn_w      = (const float*)d_in[5];
  const float* kn_w      = (const float*)d_in[6];
  const float* q_w       = (const float*)d_in[7];
  const float* k_w       = (const float*)d_in[8];
  const float* v_w       = (const float*)d_in[9];
  const float* o_w       = (const float*)d_in[10];
  const float* router_w  = (const float*)d_in[11];
  const float* gate_up   = (const float*)d_in[12];
  const float* down      = (const float*)d_in[13];
  float* out = (float*)d_out;

  unsigned char* ws = (unsigned char*)d_ws;
  const size_t o_h1   = 0;
  const size_t o_wqkv = o_h1   + (size_t)TT * EMB * 2;
  const size_t o_wo   = o_wqkv + (size_t)QKV_N * EMB * 2;
  const size_t o_wgu  = o_wo   + (size_t)EMB * (NH * HD) * 2;
  const size_t o_wdn  = o_wgu  + (size_t)NEXP * 2 * MOEH * EMB * 2;
  const size_t o_qkv  = o_wdn  + (size_t)NEXP * EMB * MOEH * 2;   // T*3072 f32
  const size_t o_qr   = o_qkv  + (size_t)TT * QKV_N * 4;
  const size_t o_kr   = o_qr   + (size_t)BB * NH * SS * HD * 2;
  const size_t o_vt   = o_kr   + (size_t)BB * NKV * SS * HD * 2;
  const size_t o_ctx  = o_vt   + (size_t)BB * NKV * SS * HD * 2;
  const size_t o_h2   = o_ctx  + (size_t)TT * NH * HD * 2;
  const size_t o_cnt  = o_h2   + (size_t)TT * EMB * 2;
  const size_t o_tok  = o_cnt  + 256;
  const size_t o_wtl  = o_tok  + (size_t)NEXP * TT * 4;
  const size_t o_act  = o_qkv;                 // alias dead qkv_f32 + Qr regions

  unsigned short* h1   = (unsigned short*)(ws + o_h1);
  unsigned short* wqkv = (unsigned short*)(ws + o_wqkv);
  unsigned short* wo   = (unsigned short*)(ws + o_wo);
  unsigned short* wgu  = (unsigned short*)(ws + o_wgu);
  unsigned short* wdn  = (unsigned short*)(ws + o_wdn);
  float*          qkvf = (float*)(ws + o_qkv);
  unsigned short* Qr   = (unsigned short*)(ws + o_qr);
  unsigned short* Kr   = (unsigned short*)(ws + o_kr);
  unsigned short* Vt   = (unsigned short*)(ws + o_vt);
  unsigned short* ctx  = (unsigned short*)(ws + o_ctx);
  unsigned short* h2   = (unsigned short*)(ws + o_h2);
  int*            cnt  = (int*)(ws + o_cnt);
  int*            tokl = (int*)(ws + o_tok);
  float*          wtl  = (float*)(ws + o_wtl);
  unsigned short* act  = (unsigned short*)(ws + o_act);

  // 1) weights fp32 -> bf16 (stay hot in 192 MB L2)
  cvt_f32_bf16<<<(2048 * EMB) / 256, 256, 0, stream>>>(q_w, wqkv, 2048 * EMB);
  cvt_f32_bf16<<<(512 * EMB) / 256, 256, 0, stream>>>(k_w, wqkv + (size_t)2048 * EMB, 512 * EMB);
  cvt_f32_bf16<<<(512 * EMB) / 256, 256, 0, stream>>>(v_w, wqkv + (size_t)2560 * EMB, 512 * EMB);
  cvt_f32_bf16<<<(EMB * 2048) / 256, 256, 0, stream>>>(o_w, wo, EMB * 2048);
  cvt_f32_bf16<<<(NEXP * 2 * MOEH * EMB) / 256, 256, 0, stream>>>(gate_up, wgu, NEXP * 2 * MOEH * EMB);
  cvt_f32_bf16<<<(NEXP * EMB * MOEH) / 256, 256, 0, stream>>>(down, wdn, NEXP * EMB * MOEH);

  // 2) rmsnorm1 -> h1 (bf16)
  rmsnorm_kernel<<<TT, 256, 0, stream>>>(x, norm1_w, h1);

  // 3) fused QKV GEMM (64x32 per wave): [T,1024] x [3072,1024]^T -> qkvf
  {
    int tiles = (TT / 64) * (QKV_N / 32);      // 6144
    gemm_rb_kernel<4, 2><<<tiles / 8, 256, 0, stream>>>(
        h1, EMB, wqkv, EMB, qkvf, QKV_N, TT, QKV_N, EMB, nullptr);
  }

  // 4) head norms + RoPE; V transpose
  qknorm_rope_kernel<<<(TT * NH) / 8, 256, 0, stream>>>(qkvf, 0,    qn_w, pos_ids, Qr, NH);
  qknorm_rope_kernel<<<(TT * NKV) / 8, 256, 0, stream>>>(qkvf, 2048, kn_w, pos_ids, Kr, NKV);
  v_transpose_kernel<<<(TT * NKV * HD) / 256, 256, 0, stream>>>(qkvf, Vt);

  // 5) flash attention -> ctx (bf16 [T, 2048])
  attn_kernel<<<BB * NH * (SS / 16), 32, 0, stream>>>(Qr, Kr, Vt, ctx);

  // 6) O projection + residual -> d_out (fp32)
  {
    int tiles = (TT / 64) * (EMB / 32);        // 2048
    gemm_rb_kernel<4, 2><<<tiles / 8, 256, 0, stream>>>(
        ctx, NH * HD, wo, NH * HD, out, EMB, TT, EMB, NH * HD, x);
  }

  // 7) rmsnorm2 + router top-2 + expert lists
  zero_counts<<<1, 32, 0, stream>>>(cnt);
  rms2_router_kernel<<<TT, 256, 0, stream>>>(out, norm2_w, router_w, h2, cnt, tokl, wtl);

  // 8) MoE: gathered gate_up (+SiLU*up), then TDM-pipelined down (+scatter)
  {
    int waves_gu = NEXP * (TT / 32) * (MOEH / 16);   // 65536
    moe_gateup_kernel<<<waves_gu / 8, 256, 0, stream>>>(h2, wgu, cnt, tokl, act);
    int waves_dn = NEXP * (TT / 32) * (EMB / 32);    // 32768
    moe_down_kernel<<<waves_dn / 8, 256, 0, stream>>>(act, wdn, cnt, tokl, wtl, out);
  }

  (void)in_sizes; (void)n_in; (void)out_size; (void)ws_size;
}
